// AttentionHead_88356067213351
// MI455X (gfx1250) — matrix operand order repaired and checked by reading, not compiled, not run on hardware
//
#include <hip/hip_runtime.h>
#include <math.h>

#define B_  8
#define T_  2048
#define D_  2048
#define H_  128

typedef __attribute__((ext_vector_type(16))) __bf16 v16bf;
typedef __attribute__((ext_vector_type(8)))  float  v8f;
typedef unsigned short u16;

// ---- CDNA5 async global->LDS path (guarded; falls back to VGPR staging) ----
#if defined(__AMDGCN__) && defined(__has_builtin)
#  if __has_builtin(__builtin_amdgcn_global_load_async_to_lds_b128)
#    define HAVE_ASYNC_LDS 1
#  endif
#endif
#ifndef HAVE_ASYNC_LDS
#  define HAVE_ASYNC_LDS 0
#endif

#if HAVE_ASYNC_LDS
typedef __attribute__((ext_vector_type(4))) int i4v;
typedef __attribute__((address_space(1))) i4v* gp1;   // global int4*
typedef __attribute__((address_space(3))) i4v* sp3;   // LDS int4*
__device__ __forceinline__ void async_b128(const void* g, void* l) {
    __builtin_amdgcn_global_load_async_to_lds_b128((gp1)g, (sp3)l, 0, 0);
}
__device__ __forceinline__ void wait_async0() {
#  if __has_builtin(__builtin_amdgcn_s_wait_asynccnt)
    __builtin_amdgcn_s_wait_asynccnt(0);
#  else
    asm volatile("s_wait_asynccnt 0" ::: "memory");
#  endif
}
#endif

// float -> bf16 round-to-nearest-even
__device__ __forceinline__ u16 f2bf(float f) {
    unsigned int u = __float_as_uint(f);
    u += 0x7FFFu + ((u >> 16) & 1u);
    return (u16)(u >> 16);
}

union Frag {
    v16bf bf;
    u16   s[16];
    uint4 q[2];
};

__device__ __forceinline__ v8f wmma_bf16(const Frag& a, const Frag& b, v8f c) {
    // D = A(16x32 bf16) * B(32x16 bf16) + C(16x16 f32)
    return __builtin_amdgcn_wmma_f32_16x16x32_bf16(
        /*neg_a=*/false, a.bf, /*neg_b=*/false, b.bf,
        /*c_mod=*/(short)0, c, /*reuse_a=*/false, /*reuse_b=*/false);
}

// ---------------------------------------------------------------------------
// Kernel 1: W[2048,128] fp32 -> Wt[3][128][2048] bf16 (transposed for B-frags)
// ---------------------------------------------------------------------------
__global__ __launch_bounds__(256)
void wt_convert_kernel(const float* __restrict__ Wq,
                       const float* __restrict__ Wk,
                       const float* __restrict__ Wv,
                       u16* __restrict__ Wt) {
    int idx = blockIdx.x * 256 + threadIdx.x;          // 3*128*2048 total
    int mat = idx / (H_ * D_);
    int rem = idx % (H_ * D_);
    int h   = rem / D_;
    int d   = rem % D_;
    const float* W = (mat == 0) ? Wq : (mat == 1) ? Wk : Wv;
    Wt[idx] = f2bf(W[(size_t)d * H_ + h]);
}

// ---------------------------------------------------------------------------
// Kernel 2: QKV projection. grid = (B*T/64, 3). block = 128 (4 waves).
// Each wave owns a 16-row M tile; N = 128 (8 tiles of 16); K-loop step 32.
// ---------------------------------------------------------------------------
__global__ __launch_bounds__(128)
void qkv_proj_kernel(const float* __restrict__ x,
                     const u16*  __restrict__ Wt,     // [3][128][2048] bf16
                     const float* __restrict__ bq,
                     const float* __restrict__ bk,
                     const float* __restrict__ bv,
                     u16* __restrict__ qkv) {         // [3][B*T][128] bf16
    __shared__ u16 Xs[64][40];     // 64x32 bf16 tile, +8 pad (rows 80B)
    __shared__ u16 Ws[128][40];    // 128x32 bf16 tile, +8 pad

    const int tid  = threadIdx.x;
    const int wid  = tid >> 5;
    const int lane = tid & 31;
    const int lh   = lane >> 4;    // half-wave select
    const int ln   = lane & 15;
    const int mat  = blockIdx.y;
    const size_t r0 = (size_t)blockIdx.x * 64;

    const u16*  W    = Wt + (size_t)mat * H_ * D_;
    const float* bias = (mat == 0) ? bq : (mat == 1) ? bk : bv;

    v8f acc[8] = {};

    for (int k0 = 0; k0 < D_; k0 += 32) {
        __syncthreads();
        // stage x tile: 64 rows x 32 cols, fp32 -> bf16 (2 threads per row)
        {
            const int row = tid >> 1, seg = tid & 1;
            const float4* s4 = (const float4*)(x + (r0 + row) * D_ + k0 + seg * 16);
            float4 f0 = s4[0], f1 = s4[1], f2 = s4[2], f3 = s4[3];
            Frag t;
            t.s[0]  = f2bf(f0.x); t.s[1]  = f2bf(f0.y); t.s[2]  = f2bf(f0.z); t.s[3]  = f2bf(f0.w);
            t.s[4]  = f2bf(f1.x); t.s[5]  = f2bf(f1.y); t.s[6]  = f2bf(f1.z); t.s[7]  = f2bf(f1.w);
            t.s[8]  = f2bf(f2.x); t.s[9]  = f2bf(f2.y); t.s[10] = f2bf(f2.z); t.s[11] = f2bf(f2.w);
            t.s[12] = f2bf(f3.x); t.s[13] = f2bf(f3.y); t.s[14] = f2bf(f3.z); t.s[15] = f2bf(f3.w);
            *(uint4*)&Xs[row][seg * 16]     = t.q[0];
            *(uint4*)&Xs[row][seg * 16 + 8] = t.q[1];
        }
        // stage W tile: 128 rows (output cols) x 32 K, one row per thread
        {
            const u16* src = W + (size_t)tid * D_ + k0;
#if HAVE_ASYNC_LDS
            #pragma unroll
            for (int j = 0; j < 4; ++j)
                async_b128(src + j * 8, &Ws[tid][j * 8]);
            wait_async0();
#else
            const uint4* s4 = (const uint4*)src;
            uint4 a0 = s4[0], a1 = s4[1], a2 = s4[2], a3 = s4[3];
            *(uint4*)&Ws[tid][0]  = a0;
            *(uint4*)&Ws[tid][8]  = a1;
            *(uint4*)&Ws[tid][16] = a2;
            *(uint4*)&Ws[tid][24] = a3;
#endif
        }
        __syncthreads();

        // A fragment: lane = row M (ln), K chunks [lh*8,+8) and [16+lh*8,+8)
        Frag a;
        const u16* xrow = &Xs[wid * 16 + ln][0];
        a.q[0] = *(const uint4*)(xrow + lh * 8);
        a.q[1] = *(const uint4*)(xrow + 16 + lh * 8);

        // preload all 8 B fragments into distinct regs, then issue 8 WMMAs
        Frag bfr[8];
        #pragma unroll
        for (int nt = 0; nt < 8; ++nt) {
            const u16* wrow = &Ws[nt * 16 + ln][0];
            bfr[nt].q[0] = *(const uint4*)(wrow + lh * 8);
            bfr[nt].q[1] = *(const uint4*)(wrow + 16 + lh * 8);
        }
        #pragma unroll
        for (int nt = 0; nt < 8; ++nt)
            acc[nt] = wmma_bf16(a, bfr[nt], acc[nt]);
    }

    // bias add + bf16 store: C layout row = wid*16 + lh*8 + r, col = nt*16+ln
    #pragma unroll
    for (int nt = 0; nt < 8; ++nt) {
        const int   col = nt * 16 + ln;
        const float bb  = bias[col];
        u16* dst = qkv + ((size_t)mat * (B_ * (size_t)T_) + r0 + wid * 16 + lh * 8) * H_ + col;
        #pragma unroll
        for (int r = 0; r < 8; ++r)
            dst[(size_t)r * H_] = f2bf(acc[nt][r] + bb);
    }
}

// ---------------------------------------------------------------------------
// Kernel 3: causal flash attention. grid = (T/64, B). block = 128 (4 waves).
// Each wave owns 16 query rows; K/V tiles of 32 keys staged per block.
// ---------------------------------------------------------------------------
__global__ __launch_bounds__(128)
void flash_attn_kernel(const u16* __restrict__ qkv, float* __restrict__ out) {
    __shared__ u16 Qs[64][136];    // 64x128 bf16 Q tile (+8 pad)
    __shared__ u16 Ks[32][136];    // 32x128 bf16 K tile (+8 pad)
    __shared__ u16 Vt[128][40];    // V tile transposed: [dim][key] (+8 pad)
    __shared__ u16 Ps[4][16][40];  // per-wave P tile 16x32 bf16 (+8 pad)

    const int tid  = threadIdx.x;
    const int wid  = tid >> 5;
    const int lane = tid & 31;
    const int lh   = lane >> 4;
    const int ln   = lane & 15;
    const int q0   = blockIdx.x * 64;
    const int b    = blockIdx.y;
    const size_t NR = (size_t)B_ * T_;

    const u16* Qg = qkv + 0 * NR * H_ + ((size_t)b * T_ + q0) * H_;
    const u16* Kg = qkv + 1 * NR * H_ + (size_t)b * T_ * H_;
    const u16* Vg = qkv + 2 * NR * H_ + (size_t)b * T_ * H_;

    // stage Q tile once (2 threads per row, 64 shorts each)
    {
        const int row = tid >> 1, seg = tid & 1;
        const u16* src = Qg + (size_t)row * H_ + seg * 64;
#if HAVE_ASYNC_LDS
        #pragma unroll
        for (int j = 0; j < 8; ++j)
            async_b128(src + j * 8, &Qs[row][seg * 64 + j * 8]);
#else
        const uint4* s4 = (const uint4*)src;
        #pragma unroll
        for (int j = 0; j < 8; ++j)
            *(uint4*)&Qs[row][seg * 64 + j * 8] = s4[j];
#endif
    }

    v8f   O[8] = {};
    float rmax[8], lsum[8];
    #pragma unroll
    for (int r = 0; r < 8; ++r) { rmax[r] = -__builtin_inff(); lsum[r] = 0.f; }

    const float scale = 0.08838834764831845f;   // 1/sqrt(128)
    const float L2E   = 1.4426950408889634f;

    const int nkt = (q0 + 64) / 32;             // causal: keys <= q0+63
    for (int kt = 0; kt < nkt; ++kt) {
        const int kbase = kt * 32;
        // stage K tile row-major (4 threads per key row) -- async path
        {
            const int row = tid >> 2, qr = tid & 3;
            const u16* src = Kg + ((size_t)kbase + row) * H_ + qr * 32;
#if HAVE_ASYNC_LDS
            #pragma unroll
            for (int j = 0; j < 4; ++j)
                async_b128(src + j * 8, &Ks[row][qr * 32 + j * 8]);
#else
            const uint4* s4 = (const uint4*)src;
            #pragma unroll
            for (int j = 0; j < 4; ++j)
                *(uint4*)&Ks[row][qr * 32 + j * 8] = s4[j];
#endif
        }
        // stage V tile transposed: Vt[dim][key] (needs VGPR round-trip)
        {
            const int kk = tid >> 2, dbase = (tid & 3) * 32;
            union { uint4 q[4]; u16 s[32]; } vb;
            const uint4* src = (const uint4*)(Vg + ((size_t)kbase + kk) * H_ + dbase);
            vb.q[0] = src[0]; vb.q[1] = src[1]; vb.q[2] = src[2]; vb.q[3] = src[3];
            #pragma unroll
            for (int j = 0; j < 32; ++j)
                Vt[dbase + j][kk] = vb.s[j];
        }
        // prefetch next tile's K/V rows into cache while we compute
        if (kt + 1 < nkt) {
            const int row = tid >> 2, qr = tid & 3;
            __builtin_prefetch(Kg + ((size_t)kbase + 32 + row) * H_ + qr * 32, 0, 1);
            __builtin_prefetch(Vg + ((size_t)kbase + 32 + row) * H_ + qr * 32, 0, 1);
        }
#if HAVE_ASYNC_LDS
        wait_async0();          // covers Q tile (iter 0) + this K tile
#endif
        __syncthreads();

        // S(16x32) = Q_tile(16x128) @ K_tile^T(128x32): two N tiles, 4 K steps
        v8f s0 = {}, s1 = {};
        #pragma unroll
        for (int kk = 0; kk < 4; ++kk) {
            Frag a;
            const u16* qrow = &Qs[wid * 16 + ln][kk * 32];
            a.q[0] = *(const uint4*)(qrow + lh * 8);
            a.q[1] = *(const uint4*)(qrow + 16 + lh * 8);
            Frag b0, b1;
            const u16* k0r = &Ks[ln][kk * 32];
            const u16* k1r = &Ks[16 + ln][kk * 32];
            b0.q[0] = *(const uint4*)(k0r + lh * 8);
            b0.q[1] = *(const uint4*)(k0r + 16 + lh * 8);
            b1.q[0] = *(const uint4*)(k1r + lh * 8);
            b1.q[1] = *(const uint4*)(k1r + 16 + lh * 8);
            s0 = wmma_bf16(a, b0, s0);
            s1 = wmma_bf16(a, b1, s1);
        }

        // masked online softmax; row m lives across 16 lanes of one half-wave
        const int rowg = q0 + wid * 16 + lh * 8;
        const int c0 = kbase + ln;
        const int c1 = kbase + 16 + ln;
        float corr[8];
        #pragma unroll
        for (int r = 0; r < 8; ++r) {
            float v0 = (c0 <= rowg + r) ? s0[r] * scale : -__builtin_inff();
            float v1 = (c1 <= rowg + r) ? s1[r] * scale : -__builtin_inff();
            float m = fmaxf(v0, v1);
            m = fmaxf(m, __shfl_xor(m, 1));
            m = fmaxf(m, __shfl_xor(m, 2));
            m = fmaxf(m, __shfl_xor(m, 4));
            m = fmaxf(m, __shfl_xor(m, 8));
            float nm = fmaxf(rmax[r], m);
            corr[r] = (nm == -__builtin_inff()) ? 1.f
                      : exp2f((rmax[r] - nm) * L2E);
            rmax[r] = nm;
            float p0 = (v0 == -__builtin_inff()) ? 0.f : exp2f((v0 - nm) * L2E);
            float p1 = (v1 == -__builtin_inff()) ? 0.f : exp2f((v1 - nm) * L2E);
            float rs = p0 + p1;
            rs += __shfl_xor(rs, 1);
            rs += __shfl_xor(rs, 2);
            rs += __shfl_xor(rs, 4);
            rs += __shfl_xor(rs, 8);
            lsum[r] = lsum[r] * corr[r] + rs;
            Ps[wid][lh * 8 + r][ln]      = f2bf(p0);
            Ps[wid][lh * 8 + r][16 + ln] = f2bf(p1);
        }

        // O = O*corr + P(16x32) @ V(32x128)
        #pragma unroll
        for (int nt = 0; nt < 8; ++nt) {
            #pragma unroll
            for (int r = 0; r < 8; ++r) O[nt][r] *= corr[r];
        }
        Frag pa;
        const u16* prow = &Ps[wid][ln][0];
        pa.q[0] = *(const uint4*)(prow + lh * 8);
        pa.q[1] = *(const uint4*)(prow + 16 + lh * 8);
        // preload all 8 V B-fragments, then issue 8 WMMAs back-to-back
        Frag bv_[8];
        #pragma unroll
        for (int nt = 0; nt < 8; ++nt) {
            const u16* vrow = &Vt[nt * 16 + ln][0];
            bv_[nt].q[0] = *(const uint4*)(vrow + lh * 8);
            bv_[nt].q[1] = *(const uint4*)(vrow + 16 + lh * 8);
        }
        #pragma unroll
        for (int nt = 0; nt < 8; ++nt)
            O[nt] = wmma_bf16(pa, bv_[nt], O[nt]);
        __syncthreads();
    }

    // normalize and store fp32 output
    float* orow = out + ((size_t)b * T_ + q0 + wid * 16 + lh * 8) * H_;
    #pragma unroll
    for (int r = 0; r < 8; ++r) {
        const float inv = 1.0f / lsum[r];
        #pragma unroll
        for (int nt = 0; nt < 8; ++nt)
            orow[(size_t)r * H_ + nt * 16 + ln] = O[nt][r] * inv;
    }
}

// ---------------------------------------------------------------------------
extern "C" void kernel_launch(void* const* d_in, const int* in_sizes, int n_in,
                              void* d_out, int out_size, void* d_ws, size_t ws_size,
                              hipStream_t stream) {
    const float* x  = (const float*)d_in[0];
    const float* Wq = (const float*)d_in[1];
    const float* bq = (const float*)d_in[2];
    const float* Wk = (const float*)d_in[3];
    const float* bk = (const float*)d_in[4];
    const float* Wv = (const float*)d_in[5];
    const float* bv = (const float*)d_in[6];
    float* out = (float*)d_out;

    // workspace layout: Wt bf16 [3][128][2048] (1.5MB) + QKV bf16 [3][B*T][128] (12MB)
    u16* Wt  = (u16*)d_ws;
    u16* QKV = Wt + (size_t)3 * H_ * D_;

    wt_convert_kernel<<<(3 * H_ * D_) / 256, 256, 0, stream>>>(Wq, Wk, Wv, Wt);
    qkv_proj_kernel<<<dim3((B_ * T_) / 64, 3), 128, 0, stream>>>(x, Wt, bq, bk, bv, QKV);
    flash_attn_kernel<<<dim3(T_ / 64, B_), 128, 0, stream>>>(QKV, out);
}
